// MLPPredictor_1554778161753
// MI455X (gfx1250) — matrix operand order repaired and checked by reading, not compiled, not run on hardware
//
#include <hip/hip_runtime.h>

// ---------------------------------------------------------------------------
// Edge MLP (GNN link predictor) for MI455X / gfx1250.
//   score[e] = relu(relu([h[src]|h[dst]] @ W1 + b1) @ W2 + b2) @ W3 + b3
// All layers computed TRANSPOSED:  OUT^T = W^T @ IN^T  with bf16 WMMA
// (v_wmma_f32_16x16x32_bf16):
//   - A operand = W^T fragments, prepacked, REGISTER-RESIDENT (192 VGPR/wave)
//   - B operand = activations, read as 32B contiguous row runs from LDS
//   - D: each lane holds 8 consecutive output features of one edge
//        -> relu + v_cvt_pk_bf16_f32 + ONE ds_store_b128 per tile epilogue,
//        -> layer-3 writes 2 coalesced global_store_b128 of f32 per lane.
// Biases live in SGPRs (readfirstlane) and are selected per lane-half with
// v_cndmask at accumulator init. sA2 aliases sX (dead after layer 1), so the
// workgroup needs only 132KB LDS -> two workgroups per 320KB WGP.
// B-fragment LDS loads double-buffered one K-step ahead of the WMMAs.
// Persistent workgroups, 128 edges per tile, 8 waves.
// ---------------------------------------------------------------------------

typedef __attribute__((ext_vector_type(16))) __bf16 v16bf;
typedef __attribute__((ext_vector_type(8)))  __bf16 bf16x8;
typedef __attribute__((ext_vector_type(8)))  float  v8f;

#define N_NODES 50000
#define N_EDGES 500000
#define D_FEAT  128
#define OUTC    16

#define TILE_E  128                 // edges per workgroup tile
#define WAVES   8
#define THREADS (WAVES * 32)
#define N_TILES ((N_EDGES + TILE_E - 1) / TILE_E)   // 3907

// LDS row strides in bf16 elements, padded (+8) to spread banks
#define SX_STRIDE  264              // 256 + 8
#define SA2_STRIDE 136              // 128 + 8
// sX and sA1 each 128*264*2 = 67584 B; sA2 (34816 B) aliases sX.
#define SMEM_BYTES (2 * 128 * SX_STRIDE * 2)        // 135168 B

// Packed-weight workspace layout (bf16 elements; 512 elems = 1 fragment)
#define W1P_OFF   0
#define W1P_FRAGS (8 * 16)          // ks * t
#define W2P_OFF   (W1P_FRAGS * 512)
#define W2P_FRAGS (8 * 8)
#define W3P_OFF   (W2P_OFF + W2P_FRAGS * 512)
#define W3P_FRAGS 4
#define PACK_TOTAL (W3P_OFF + W3P_FRAGS * 512)     // 100352 elems = 200704 B

// ---------------------------------------------------------------------------
// Prepack: fp32 W [K_in][N_out] -> bf16 A-layout fragments of W^T.
// A 16x32 layout: lane: M = lane%16 (= output dim within tile);
// element e: K = e + 16*(lane>=16)  (runs {0-15}/{16-31}): contiguous k order chosen here, B uses the same.
// Fragment f=(ks,t) stored as 32 lanes x 16 bf16 contiguous (1KB).
// ---------------------------------------------------------------------------
__global__ void prepack_weights(const float* __restrict__ W1,
                                const float* __restrict__ W2,
                                const float* __restrict__ W3,
                                __bf16* __restrict__ pack) {
    int gid = blockIdx.x * blockDim.x + threadIdx.x;
    if (gid >= PACK_TOTAL) return;
    const float* W; int Ndim, off;
    if (gid < W2P_OFF)      { W = W1; Ndim = 256; off = gid; }
    else if (gid < W3P_OFF) { W = W2; Ndim = 128; off = gid - W2P_OFF; }
    else                    { W = W3; Ndim = 16;  off = gid - W3P_OFF; }
    int frag   = off >> 9;          // /512
    int within = off & 511;
    int lane   = within >> 4;
    int e      = within & 15;
    int tcnt   = Ndim >> 4;
    int ks     = frag / tcnt;
    int t      = frag % tcnt;
    int kl = e + ((lane >= 16) ? 16 : 0);
    int k  = ks * 32 + kl;
    int m  = t * 16 + (lane & 15);  // output dim
    pack[gid] = (__bf16)W[k * Ndim + m];
}

// Prepacked W^T A-fragment: lane reads its contiguous 32 bytes.
__device__ __forceinline__ v16bf load_w_frag(const __bf16* pack, int frag, int lane) {
    return *(const v16bf*)(pack + frag * 512 + lane * 16);
}

// Activation B-fragment (K=32 x N=16) from row-major LDS buffer.
// lane: N = lane%16 (edge row), elements: K = kbase + 16*laneHi + e
// -> one 32B contiguous run = two ds_load_b128.
__device__ __forceinline__ v16bf load_b_act(const __bf16* buf, int edgeRow,
                                            int stride, int kbase, int laneHi) {
    const __bf16* p = buf + edgeRow * stride + kbase + (laneHi ? 16 : 0);
    union { v16bf v; bf16x8 h[2]; } u;
    u.h[0] = *(const bf16x8*)(p);
    u.h[1] = *(const bf16x8*)(p + 8);
    return u.v;
}

__device__ __forceinline__ float relu(float x) { return x > 0.0f ? x : 0.0f; }

// Wave-uniform scalar load: value pinned to an SGPR via readfirstlane.
__device__ __forceinline__ float uload(const float* p) {
    union { float f; int i; } u; u.f = *p;
    u.i = __builtin_amdgcn_readfirstlane(u.i);
    return u.f;
}

__global__ void __launch_bounds__(THREADS, 2)
edge_mlp(const float* __restrict__ h,
         const long long* __restrict__ srcIdx,
         const long long* __restrict__ dstIdx,
         const float* __restrict__ b1,
         const float* __restrict__ b2,
         const float* __restrict__ b3,
         const __bf16* __restrict__ pack,
         float* __restrict__ out) {
    extern __shared__ __bf16 smem[];
    __bf16* sX  = smem;                      // [128][SX_STRIDE]  concat input
    __bf16* sA1 = smem + 128 * SX_STRIDE;    // [128][SX_STRIDE]  act1 (256 wide)
    __bf16* sA2 = smem;                      // [128][SA2_STRIDE] act2 — ALIASES sX

    const int lane   = threadIdx.x & 31;
    const int wave   = threadIdx.x >> 5;
    const int laneHi = lane >> 4;
    const int lane15 = lane & 15;

    // ---- register-resident W^T fragments (persistent across all tiles) ----
    const int t0 = 2 * wave, t1 = 2 * wave + 1;   // layer-1 owned out-tiles
    v16bf A1r[16];
#pragma unroll
    for (int ks = 0; ks < 8; ++ks) {
        A1r[2 * ks + 0] = load_w_frag(pack + W1P_OFF, ks * 16 + t0, lane);
        A1r[2 * ks + 1] = load_w_frag(pack + W1P_OFF, ks * 16 + t1, lane);
    }
    v16bf A2r[8];                                 // layer-2 owned out-tile = wave
#pragma unroll
    for (int ks = 0; ks < 8; ++ks)
        A2r[ks] = load_w_frag(pack + W2P_OFF, ks * 8 + wave, lane);

    const int rbase = laneHi ? 8 : 0;             // out-dim offset of this lane half

    // ---- biases in SGPRs (values differ only by r and lane half) ----
    float b1L0[8], b1H0[8], b1L1[8], b1H1[8], b2L[8], b2H[8], b3L[8], b3H[8];
#pragma unroll
    for (int r = 0; r < 8; ++r) {
        b1L0[r] = uload(b1 + t0 * 16 + r);      b1H0[r] = uload(b1 + t0 * 16 + 8 + r);
        b1L1[r] = uload(b1 + t1 * 16 + r);      b1H1[r] = uload(b1 + t1 * 16 + 8 + r);
        b2L[r]  = uload(b2 + wave * 16 + r);    b2H[r]  = uload(b2 + wave * 16 + 8 + r);
        b3L[r]  = uload(b3 + r);                b3H[r]  = uload(b3 + 8 + r);
    }

    for (int tile = blockIdx.x; tile < N_TILES; tile += gridDim.x) {
        const int base = tile * TILE_E;

        // layer-3 W^T pointer made loop-variant so its 4 fragments are not
        // hoisted into permanently-live VGPRs (layer 3 is ~2% of the FLOPs).
        const __bf16* w3p = pack + W3P_OFF;
        asm volatile("" : "+s"(w3p));

        // ---- stage X = [h[src] | h[dst]] as bf16 into LDS (row-major) ----
        {
            int row  = threadIdx.x >> 1;     // 0..127
            int part = threadIdx.x & 1;      // 0 = src half, 1 = dst half
            int e = base + row; if (e >= N_EDGES) e = N_EDGES - 1;
            long long idx = part ? dstIdx[e] : srcIdx[e];
            const float4* hp = (const float4*)(h + idx * (long long)D_FEAT);
            __bf16* dp = sX + row * SX_STRIDE + part * 128;
#pragma unroll 4
            for (int j = 0; j < 32; ++j) {
                float4 v = hp[j];
                union { __bf16 b[4]; unsigned long long u; } q;
                q.b[0] = (__bf16)v.x; q.b[1] = (__bf16)v.y;
                q.b[2] = (__bf16)v.z; q.b[3] = (__bf16)v.w;
                *(unsigned long long*)(dp + j * 4) = q.u;   // ds_store_b64
            }
        }
        __syncthreads();

        // ---- layer 1: act1^T tiles; wave owns out-dims [32w, 32w+32) ----
#pragma unroll 1
        for (int et = 0; et < 8; ++et) {
            v8f acc0, acc1;
#pragma unroll
            for (int r = 0; r < 8; ++r) {        // v_cndmask from SGPR biases
                acc0[r] = laneHi ? b1H0[r] : b1L0[r];
                acc1[r] = laneHi ? b1H1[r] : b1L1[r];
            }
            const int erow = et * 16 + lane15;
            v16bf B = load_b_act(sX, erow, SX_STRIDE, 0, laneHi);
#pragma unroll
            for (int ks = 0; ks < 8; ++ks) {
                v16bf Bn = (ks < 7) ? load_b_act(sX, erow, SX_STRIDE, (ks + 1) * 32, laneHi)
                                    : B;          // prefetch next K-step
                acc0 = __builtin_amdgcn_wmma_f32_16x16x32_bf16(
                           false, A1r[2 * ks + 0], false, B, (short)0, acc0, false, false);
                acc1 = __builtin_amdgcn_wmma_f32_16x16x32_bf16(
                           false, A1r[2 * ks + 1], false, B, (short)0, acc1, false, false);
                B = Bn;
            }
            // lane holds 8 consecutive output features of edge `erow`
            bf16x8 o0, o1;
#pragma unroll
            for (int r = 0; r < 8; ++r) {
                o0[r] = (__bf16)relu(acc0[r]);
                o1[r] = (__bf16)relu(acc1[r]);
            }
            __bf16* dp = sA1 + erow * SX_STRIDE + rbase;
            *(bf16x8*)(dp + t0 * 16) = o0;       // ds_store_b128
            *(bf16x8*)(dp + t1 * 16) = o1;       // ds_store_b128
        }
        __syncthreads();

        // ---- layer 2: act2^T tiles; wave owns out-dims [16w, 16w+16) ----
#pragma unroll 1
        for (int et = 0; et < 8; ++et) {
            v8f acc;
#pragma unroll
            for (int r = 0; r < 8; ++r)
                acc[r] = laneHi ? b2H[r] : b2L[r];
            const int erow = et * 16 + lane15;
            v16bf B = load_b_act(sA1, erow, SX_STRIDE, 0, laneHi);
#pragma unroll
            for (int ks = 0; ks < 8; ++ks) {
                v16bf Bn = (ks < 7) ? load_b_act(sA1, erow, SX_STRIDE, (ks + 1) * 32, laneHi)
                                    : B;
                acc = __builtin_amdgcn_wmma_f32_16x16x32_bf16(
                          false, A2r[ks], false, B, (short)0, acc, false, false);
                B = Bn;
            }
            bf16x8 o;
#pragma unroll
            for (int r = 0; r < 8; ++r) o[r] = (__bf16)relu(acc[r]);
            *(bf16x8*)(sA2 + erow * SA2_STRIDE + wave * 16 + rbase) = o;  // ds_store_b128
        }
        __syncthreads();

        // ---- layer 3: wave w handles edge-tile w (16 edges), K = 128 ----
        {
            v8f acc;
#pragma unroll
            for (int r = 0; r < 8; ++r)
                acc[r] = laneHi ? b3H[r] : b3L[r];
            const int erow = wave * 16 + lane15;
            v16bf B = load_b_act(sA2, erow, SA2_STRIDE, 0, laneHi);
#pragma unroll
            for (int ks = 0; ks < 4; ++ks) {
                v16bf Bn = (ks < 3) ? load_b_act(sA2, erow, SA2_STRIDE, (ks + 1) * 32, laneHi)
                                    : B;
                v16bf A = load_w_frag(w3p, ks, lane);
                acc = __builtin_amdgcn_wmma_f32_16x16x32_bf16(
                          false, A, false, B, (short)0, acc, false, false);
                B = Bn;
            }
            // lane writes 8 consecutive f32 classes of one edge: 2x b128 stores
            int e = base + erow;
            if (e < N_EDGES) {
                float4 lo, hi;
                lo.x = acc[0]; lo.y = acc[1]; lo.z = acc[2]; lo.w = acc[3];
                hi.x = acc[4]; hi.y = acc[5]; hi.z = acc[6]; hi.w = acc[7];
                float4* op = (float4*)(out + (long long)e * OUTC + rbase);
                op[0] = lo;
                op[1] = hi;
            }
        }
        __syncthreads();   // sX/sA1/sA2 reused next tile
    }
}

extern "C" void kernel_launch(void* const* d_in, const int* in_sizes, int n_in,
                              void* d_out, int out_size, void* d_ws, size_t ws_size,
                              hipStream_t stream) {
    const float*     h   = (const float*)d_in[0];
    const long long* src = (const long long*)d_in[1];
    const long long* dst = (const long long*)d_in[2];
    const float*     W1  = (const float*)d_in[3];
    const float*     b1  = (const float*)d_in[4];
    const float*     W2  = (const float*)d_in[5];
    const float*     b2  = (const float*)d_in[6];
    const float*     W3  = (const float*)d_in[7];
    const float*     b3  = (const float*)d_in[8];
    __bf16* pack = (__bf16*)d_ws;                   // 200704 bytes used
    float*  out  = (float*)d_out;

    prepack_weights<<<(PACK_TOTAL + 255) / 256, 256, 0, stream>>>(W1, W2, W3, pack);

    int grid = 512;
    if (grid > N_TILES) grid = N_TILES;
    edge_mlp<<<grid, THREADS, SMEM_BYTES, stream>>>(h, src, dst, b1, b2, b3,
                                                    (const __bf16*)pack, out);
}
